// Soft_GSL_Block_PE_54829552501125
// MI455X (gfx1250) — matrix operand order repaired and checked by reading, not compiled, not run on hardware
//
#include <hip/hip_runtime.h>

// CDNA5 / gfx1250 fused Soft-GSL block, v3.
// Stage 1: all four input projections fused into one 160x8 GEMM on
//          v_wmma_f32_16x16x32_f16 (bias via K=6 ones row); wave-scalar tile
//          indices (readfirstlane) -> uniform branches; A operand selected by
//          address (zero-row trick); pos/attn scatters packed to ds_store_b128.
// Stage 2: pos_enc (64x64, two k-steps) + attn logits (64x16, pos_enc as C
//          operand) on WMMA; softmax/max reductions over K=20 fused in LDS.

typedef __attribute__((ext_vector_type(16))) _Float16 v16h;
typedef __attribute__((ext_vector_type(8)))  _Float16 v8h;
typedef __attribute__((ext_vector_type(8)))  float    v8f;

#define CH_STRIDE 163840   // 8192 * 20
#define NPTS      320      // 16 n * 20 k per workgroup
#define EPSV      1e-5f
#define NSLOPE    0.2f

__device__ __forceinline__ float lrelu(float v) { return v > 0.f ? v : NSLOPE * v; }

__global__ __launch_bounds__(256, 1)
void soft_gsl_fused(
    const float* __restrict__ x,
    const float* __restrict__ trans_w, const float* __restrict__ trans_g,
    const float* __restrict__ trans_b, const float* __restrict__ trans_m,
    const float* __restrict__ trans_v,
    const float* __restrict__ pos_w1,  const float* __restrict__ pos_g,
    const float* __restrict__ pos_b,   const float* __restrict__ pos_m,
    const float* __restrict__ pos_v,
    const float* __restrict__ pos_w2,  const float* __restrict__ pos_b2,
    const float* __restrict__ gate_w1, const float* __restrict__ gate_g,
    const float* __restrict__ gate_b,  const float* __restrict__ gate_m,
    const float* __restrict__ gate_v,
    const float* __restrict__ gate_w2, const float* __restrict__ gate_b2,
    const float* __restrict__ attn_w1, const float* __restrict__ attn_g,
    const float* __restrict__ attn_b,  const float* __restrict__ attn_m,
    const float* __restrict__ attn_v,
    const float* __restrict__ attn_w2, const float* __restrict__ attn_b2,
    float* __restrict__ out)
{
    __shared__ _Float16 sTrans[64 * NPTS];          // h1 -> trans_feat (f16)   40 KB
    __shared__ _Float16 sAttn [64 * NPTS];          // attn_logits (f16)        40 KB
    __shared__ _Float16 sBpos [20 * 2 * 32 * 16];   // pos_h, WMMA-B layout     40 KB
    __shared__ _Float16 sBattn[20 * 32 * 16];       // attn_h, WMMA-B layout    20 KB
    __shared__ _Float16 sBx   [20 * 32 * 16];       // [x;1;0], WMMA-B layout   20 KB
    __shared__ _Float16 sGateH[16 * NPTS];          // gate_h                   10 KB
    __shared__ float    sGate[NPTS];
    __shared__ _Float16 sW1h[160 * 16];             // stage-1 A rows: 8 w + 8 zeros
    __shared__ _Float16 sZeroA[16];                 // all-zero A row (hi lanes)
    __shared__ float    sGw2[17];                   // gate_w2[16] + gate_b2

    const int tid  = threadIdx.x;
    const int lane = tid & 31;
    const int wuni = __builtin_amdgcn_readfirstlane(tid >> 5);   // scalar wave id
    const int colA = lane & 15, hi = lane >> 4;
    const int b    = blockIdx.x >> 9;               // 512 WGs per batch
    const int n0   = (blockIdx.x & 511) << 4;       // 16 n per WG
    const size_t xbase = (size_t)b * 6 * CH_STRIDE + (size_t)n0 * 20;

    // ---- Phase 0: fold BN into a unified stage-1 A matrix (160 rows x K=8, f16) ----
    // rows 0..63 trans (w0..5, bias), 64..127 pos1 (0,0,0,w0..2, bias),
    // 128..143 gate1, 144..159 attn1. K=6 column multiplies the B "ones" row.
    if (tid < 160) {
        float w[6], bias;
        if (tid < 64) {
            float sc = trans_g[tid] * rsqrtf(trans_v[tid] + EPSV);
            #pragma unroll
            for (int k = 0; k < 6; ++k) w[k] = trans_w[tid * 6 + k] * sc;
            bias = trans_b[tid] - trans_m[tid] * sc;
        } else if (tid < 128) {
            int r = tid - 64;
            float sc = pos_g[r] * rsqrtf(pos_v[r] + EPSV);
            w[0] = w[1] = w[2] = 0.f;
            #pragma unroll
            for (int k = 0; k < 3; ++k) w[3 + k] = pos_w1[r * 3 + k] * sc;
            bias = pos_b[r] - pos_m[r] * sc;
        } else if (tid < 144) {
            int r = tid - 128;
            float sc = gate_g[r] * rsqrtf(gate_v[r] + EPSV);
            #pragma unroll
            for (int k = 0; k < 6; ++k) w[k] = gate_w1[r * 6 + k] * sc;
            bias = gate_b[r] - gate_m[r] * sc;
        } else {
            int r = tid - 144;
            float sc = attn_g[r] * rsqrtf(attn_v[r] + EPSV);
            #pragma unroll
            for (int k = 0; k < 6; ++k) w[k] = attn_w1[r * 6 + k] * sc;
            bias = attn_b[r] - attn_m[r] * sc;
        }
        #pragma unroll
        for (int k = 0; k < 6; ++k) sW1h[tid * 16 + k] = (_Float16)w[k];
        sW1h[tid * 16 + 6] = (_Float16)bias;
        #pragma unroll
        for (int k = 7; k < 16; ++k) sW1h[tid * 16 + k] = (_Float16)0.f;
    } else if (tid == 160) {
        for (int k = 0; k < 16; ++k) sGw2[k] = gate_w2[k];
        sGw2[16] = gate_b2[0];
        for (int q = 0; q < 16; ++q) sZeroA[q] = (_Float16)0.f;
    }

    // zero lanes 16..31 (K=16..31 rows) of sBattn and sBx
    {
        v8h z;
        #pragma unroll
        for (int q = 0; q < 8; ++q) z[q] = (_Float16)0.f;
        for (int i = tid; i < 640; i += 256) {
            int addr = ((i >> 5) * 32 + 16) * 16 + (i & 31) * 8;
            *(v8h*)&sBattn[addr] = z;
            *(v8h*)&sBx[addr]    = z;
        }

        // ---- Phase A1': pack [x0..x5, 1, 0] into B layout (lane=col, elem j=K) ----
        for (int p = tid; p < NPTS; p += 256) {
            v8h pk;
            #pragma unroll
            for (int c = 0; c < 6; ++c) pk[c] = (_Float16)x[xbase + (size_t)c * CH_STRIDE + p];
            pk[6] = (_Float16)1.f;   // bias lane
            pk[7] = (_Float16)0.f;
            int base = (((p >> 4) * 32) + (p & 15)) << 4;
            *(v8h*)&sBx[base]     = pk;  // K = 0..7
            *(v8h*)&sBx[base + 8] = z;   // K = 8..15
        }
    }
    __syncthreads();

    // ---- Stage 1: unified 160x8 projection GEMM on WMMA, lrelu + scatter ----
    {
        int t  = wuni * 25;                 // 200 tiles = 20 ntiles x 10 mtiles
        int nt = t / 10, mt = t % 10;       // SALU-provable uniform
        v16h bX;
        #pragma unroll 1
        for (int i = 0; i < 25; ++i) {
            if (i == 0 || mt == 0)
                bX = *(const v16h*)&sBx[(nt * 32 + lane) << 4];
            // A operand: lanes 0..15 read the weight row, lanes 16..31 a zero row
            const _Float16* aptr = hi ? sZeroA : &sW1h[(mt * 16 + colA) * 16];
            v16h aS = *(const v16h*)aptr;

            v8f d = {};
            d = __builtin_amdgcn_wmma_f32_16x16x32_f16(false, aS, false, bX,
                                                       (short)0, d, false, false);
            const int n = nt * 16 + colA;
            if (mt < 4) {                                   // h1 rows 0..63
                #pragma unroll
                for (int v = 0; v < 8; ++v)
                    sTrans[(mt * 16 + v + (hi << 3)) * NPTS + n] = (_Float16)lrelu(d[v]);
            } else if (mt < 8) {                            // pos_h -> B layout (packed)
                v8h hv;
                #pragma unroll
                for (int v = 0; v < 8; ++v) hv[v] = (_Float16)lrelu(d[v]);
                int K0 = (mt - 4) * 16 + (hi << 3);          // aligned 8-block of K
                int addr = ((((nt << 1) + (K0 >> 5)) * 32 + (colA + (K0 & 16))) << 4)
                         + (K0 & 15);
                *(v8h*)&sBpos[addr] = hv;
            } else if (mt == 8) {                           // gate_h rows 0..15
                #pragma unroll
                for (int v = 0; v < 8; ++v)
                    sGateH[(v + (hi << 3)) * NPTS + n] = (_Float16)lrelu(d[v]);
            } else {                                        // attn_h -> B layout (packed)
                v8h hv;
                #pragma unroll
                for (int v = 0; v < 8; ++v) hv[v] = (_Float16)lrelu(d[v]);
                *(v8h*)&sBattn[((nt * 32 + colA) << 4) + (hi << 3)] = hv;
            }
            if (++mt == 10) { mt = 0; ++nt; }
        }
    }
    __syncthreads();

    // ---- Gate scalar per point (16 MACs) ----
    for (int p = tid; p < NPTS; p += 256) {
        float g = sGw2[16];
        #pragma unroll
        for (int c = 0; c < 16; ++c) g += sGw2[c] * (float)sGateH[c * NPTS + p];
        sGate[p] = 1.f / (1.f + __expf(-g));
    }

    // ---- Stage 2: pos_enc (64x64) + attn logits (64x16) on WMMA ----
    {
        const int mtile = wuni >> 1;                  // scalar, constant per wave
        v16h aPos0, aPos1, aAttn;
        float pb[8], ab[8];
        {
            const int row = mtile * 16 + colA;
            #pragma unroll
            for (int j = 0; j < 16; ++j) {            // A elem j -> K=j+16*hi
                int kl = j + (hi << 4);
                aPos0[j] = (_Float16)pos_w2[row * 64 + kl];
                aPos1[j] = (_Float16)pos_w2[row * 64 + 32 + kl];
                aAttn[j] = (hi == 0) ? (_Float16)attn_w2[row * 16 + j]
                                   : (_Float16)0.f;
            }
            #pragma unroll
            for (int v = 0; v < 8; ++v) {
                int m = mtile * 16 + v + hi * 8;
                pb[v] = pos_b2[m];
                ab[v] = attn_b2[m];
            }
        }

        #pragma unroll 1
        for (int i = 0; i < 10; ++i) {
            const int ntile = ((wuni & 1) * 10) + i;  // scalar

            v16h bP0 = *(const v16h*)&sBpos[(((ntile << 1) + 0) * 32 + lane) << 4];
            v16h bP1 = *(const v16h*)&sBpos[(((ntile << 1) + 1) * 32 + lane) << 4];
            v16h bA  = *(const v16h*)&sBattn[(ntile * 32 + lane) << 4];

            v8f acc = {};
            acc = __builtin_amdgcn_wmma_f32_16x16x32_f16(false, aPos0, false, bP0,
                                                         (short)0, acc, false, false);
            acc = __builtin_amdgcn_wmma_f32_16x16x32_f16(false, aPos1, false, bP1,
                                                         (short)0, acc, false, false);
            #pragma unroll
            for (int v = 0; v < 8; ++v) acc[v] += pb[v];   // pos_enc = GEMM + pos_b2
            v8f pe = acc;
            v8f al = __builtin_amdgcn_wmma_f32_16x16x32_f16(false, aAttn, false, bA,
                                                            (short)0, pe, false, false);
            // C/D layout: VGPR v, lanes<16 -> M=v, lanes>=16 -> M=v+8, N=lane%16
            #pragma unroll
            for (int v = 0; v < 8; ++v) {
                int m = mtile * 16 + v + hi * 8;
                int idx = m * NPTS + ntile * 16 + colA;
                float tf = (float)sTrans[idx] + pe[v];     // trans_feat = h1 + pos_enc
                sTrans[idx] = (_Float16)tf;
                sAttn[idx]  = (_Float16)(al[v] + ab[v]);
            }
        }
    }
    __syncthreads();

    // ---- Phase B: softmax over K=20, weighted sum + gated max ----
    for (int o = tid; o < 1024; o += 256) {        // 64 ch x 16 n
        const int c = o >> 4, n = o & 15;
        const int base = c * NPTS + n * 20;
        float l[20];
        float mx = -1e30f;
        #pragma unroll
        for (int k = 0; k < 20; ++k) {
            l[k] = (float)sAttn[base + k];
            mx = fmaxf(mx, l[k]);
        }
        float Z = 0.f, ws = 0.f, cm = -1e30f;
        #pragma unroll
        for (int k = 0; k < 20; ++k) {
            float e  = __expf(l[k] - mx);
            float tf = (float)sTrans[base + k];
            Z  += e;
            ws += tf * e;
            cm  = fmaxf(cm, tf * sGate[n * 20 + k]);
        }
        out[(size_t)b * 64 * 8192 + (size_t)c * 8192 + (n0 + n)] = ws / Z + cm;
    }
}

extern "C" void kernel_launch(void* const* d_in, const int* in_sizes, int n_in,
                              void* d_out, int out_size, void* d_ws, size_t ws_size,
                              hipStream_t stream) {
    (void)in_sizes; (void)n_in; (void)out_size; (void)d_ws; (void)ws_size;
    const float* p[27];
    for (int i = 0; i < 27; ++i) p[i] = (const float*)d_in[i];
    dim3 grid(4096), block(256);
    soft_gsl_fused<<<grid, block, 0, stream>>>(
        p[0],  p[1],  p[2],  p[3],  p[4],  p[5],
        p[6],  p[7],  p[8],  p[9],  p[10],
        p[11], p[12],
        p[13], p[14], p[15], p[16], p[17],
        p[18], p[19],
        p[20], p[21], p[22], p[23], p[24],
        p[25], p[26],
        (float*)d_out);
}
